// simpleLSTM_39221641347108
// MI455X (gfx1250) — matrix operand order repaired and checked by reading, not compiled, not run on hardware
//
#include <hip/hip_runtime.h>

// Collapsed "LSTM" for MI455X (gfx1250, wave32):
//   gx = X[B*T, I] x Wi[4H, I]^T   (M=32768, N=4096, K=1024)  -- bf16 WMMA, f32 accum
//   gh = h0[B, H]  x Wh[4H, H]^T   (M=64,    N=4096, K=1024)  -- bf16 WMMA, f32 accum
//   epilogue: i,f,g,o gates fused in-register, h/c written out.
//
// Fast path: X and Wi pre-converted to bf16 in ws (both L2-resident, 72 MB
// vs 192 MB L2); weight tiles staged through LDS (double-buffered) and shared
// by 8 waves covering different M rows -> 52 FLOP/B from L2 instead of 21.8.

typedef __attribute__((ext_vector_type(16))) __bf16 v16bf;
typedef __attribute__((ext_vector_type(8)))  __bf16 v8bf;
typedef __attribute__((ext_vector_type(8)))  float  v8f;
typedef __attribute__((ext_vector_type(4)))  float  v4f;

namespace {
constexpr int kB = 64;
constexpr int kT = 512;
constexpr int kI = 1024;
constexpr int kH = 1024;
constexpr int kN = 4096;   // 4*H
}

__device__ __forceinline__ v8f wmma_bf16(v16bf a, v16bf b, v8f c) {
  // D = A(16x32) * B(32x16) + C, f32 accumulate
  return __builtin_amdgcn_wmma_f32_16x16x32_bf16(
      /*neg_a=*/false, a, /*neg_b=*/false, b,
      /*c_mod=*/(short)0, c, /*reuse_a=*/false, /*reuse_b=*/false);
}

__device__ __forceinline__ v16bf pack16(v4f q0, v4f q1, v4f q2, v4f q3) {
  v16bf r;
#pragma unroll
  for (int j = 0; j < 4; ++j) {
    r[j]      = (__bf16)q0[j];
    r[j + 4]  = (__bf16)q1[j];
    r[j + 8]  = (__bf16)q2[j];
    r[j + 12] = (__bf16)q3[j];
  }
  return r;
}

__device__ __forceinline__ v16bf cat8(v8bf lo, v8bf hi) {
  v16bf r;
#pragma unroll
  for (int j = 0; j < 8; ++j) { r[j] = lo[j]; r[j + 8] = hi[j]; }
  return r;
}

// ---- fragment loaders (contiguous k order, same in A and B) ----
// A (16x32, 16-bit): lane holds row m=laneLo; elems e -> K=kb+laneHi*16+e.
__device__ __forceinline__ v16bf load_a_frag_f32(const float* __restrict__ row,
                                                 int kb, int laneHi) {
  const v4f* p0 = (const v4f*)(row + kb + laneHi * 16);
  const v4f* p1 = (const v4f*)(row + kb + laneHi * 16 + 8);
  return pack16(p0[0], p0[1], p1[0], p1[1]);
}
__device__ __forceinline__ v16bf load_a_frag_bf16(const __bf16* __restrict__ row,
                                                  int kb, int laneHi) {
  const v8bf* p0 = (const v8bf*)(row + kb + laneHi * 16);       // 16B load
  const v8bf* p1 = (const v8bf*)(row + kb + laneHi * 16 + 8);   // 16B load
  return cat8(p0[0], p1[0]);
}
// B (32x16, 16-bit): lane holds col n=laneLo; elems e -> K=kb+laneHi*16+e.
__device__ __forceinline__ v16bf load_b_frag_f32(const float* __restrict__ row,
                                                 int kb, int laneHi) {
  const v4f* p = (const v4f*)(row + kb + laneHi * 16);
  return pack16(p[0], p[1], p[2], p[3]);
}
__device__ __forceinline__ v16bf load_b_frag_bf16(const __bf16* __restrict__ row,
                                                  int kb, int laneHi) {
  const v8bf* p = (const v8bf*)(row + kb + laneHi * 16);
  return cat8(p[0], p[1]);
}

__device__ __forceinline__ float sigm(float x) {
  return 1.0f / (1.0f + __expf(-x));
}
__device__ __forceinline__ float tanh_(float x) {
  return 2.0f * sigm(2.0f * x) - 1.0f;
}

// ---------------- f32 -> bf16 elementwise pre-pass (8 elems/thread) --------
__global__ __launch_bounds__(256, 1)
void f32_to_bf16_kernel(const void* __restrict__ srcp, void* __restrict__ dstp,
                        int n8) {
  const int i = blockIdx.x * blockDim.x + threadIdx.x;
  if (i >= n8) return;
  const float* src = (const float*)srcp;
  __bf16*      dst = (__bf16*)dstp;
  const v4f* p = (const v4f*)(src + (size_t)i * 8);
  const v4f a = p[0], b = p[1];
  v8bf r;
#pragma unroll
  for (int j = 0; j < 4; ++j) {
    r[j]     = (__bf16)a[j];
    r[j + 4] = (__bf16)b[j];
  }
  *(v8bf*)(dst + (size_t)i * 8) = r;
}

// ---------------- gh = h0 x Wh^T : one wave per 16x16 tile ----------------
__global__ __launch_bounds__(32, 1)
void gh_gemm_kernel(const float* __restrict__ h0,
                    const float* __restrict__ Wh,
                    float* __restrict__ gh) {
  const int lane   = threadIdx.x & 31;
  const int laneLo = lane & 15;
  const int laneHi = lane >> 4;
  const int mBase  = blockIdx.x * 16;   // over B (=64)
  const int nBase  = blockIdx.y * 16;   // over 4H (=4096)

  const float* arow = h0 + (size_t)(mBase + laneLo) * kH;
  const float* wrow = Wh + (size_t)(nBase + laneLo) * kH;

  v8f acc = {0.f, 0.f, 0.f, 0.f, 0.f, 0.f, 0.f, 0.f};
  for (int kb = 0; kb < kH; kb += 32) {
    acc = wmma_bf16(load_a_frag_f32(arow, kb, laneHi),
                    load_b_frag_f32(wrow, kb, laneHi), acc);
  }
#pragma unroll
  for (int r = 0; r < 8; ++r) {
    const int m = mBase + laneHi * 8 + r;     // C/D layout: M = r + 8*(lane/16)
    gh[(size_t)m * kN + nBase + laneLo] = acc[r];
  }
}

// ---------- shared epilogue: fused 4-gate LSTM pointwise + stores ----------
__device__ __forceinline__ void lstm_epilogue(
    v8f acc[2][4], int mBase, int hBase, int laneLo, int laneHi,
    const float* __restrict__ gh, const float* __restrict__ bi,
    const float* __restrict__ bh, const float* __restrict__ cinit,
    float* __restrict__ out) {
  const int b    = mBase / kT;     // wave-uniform: 32-row tiles never cross T
  const int hcol = hBase + laneLo;

  float ghv[4], bsum[4];
#pragma unroll
  for (int g = 0; g < 4; ++g) {
    ghv[g]  = gh[(size_t)b * kN + g * kH + hcol];
    bsum[g] = bi[g * kH + hcol] + bh[g * kH + hcol];
  }
  const float c0v = cinit[(size_t)b * kH + hcol];
  float* tail = out + (size_t)kT * kB * kH;

#pragma unroll
  for (int mi = 0; mi < 2; ++mi) {
    const int m0 = mBase + mi * 16 + laneHi * 8;
#pragma unroll
    for (int r = 0; r < 8; ++r) {
      const int m = m0 + r;
      const int t = m - b * kT;
      const float pi = acc[mi][0][r] + ghv[0] + bsum[0];
      const float pf = acc[mi][1][r] + ghv[1] + bsum[1];
      const float pg = acc[mi][2][r] + ghv[2] + bsum[2];
      const float po = acc[mi][3][r] + ghv[3] + bsum[3];
      const float iv = sigm(pi);
      const float fv = sigm(pf);
      const float gv = tanh_(pg);
      const float ov = sigm(po);
      const float cv = fv * c0v + iv * gv;
      const float hv = ov * tanh_(cv);
      out[((size_t)t * kB + b) * kH + hcol] = hv;     // h[t][b][hcol]
      if (t == kT - 1) {
        tail[(size_t)b * kH + hcol] = hv;                      // h[-1:]
        tail[(size_t)kB * kH + (size_t)b * kH + hcol] = cv;    // c[-1:]
      }
    }
  }
}

// ------------- FAST PATH: bf16 operands, B tile shared via LDS -------------
// Block: 256 threads = 8 waves over M (8x32 = 256 rows), same 16 h-cols.
// Per K-step the block pulls A=16KB + B=4KB from L2 for 64 WMMAs.
// LDS: double-buffered B tile [gate][n][k] with padded k-stride (bank-clean).
__global__ __launch_bounds__(256, 1)
void lstm_bf16_lds_kernel(const void* __restrict__ Xp,   // [B*T, I] bf16
                          const void* __restrict__ Wp,   // [4H, I]  bf16
                          const float* __restrict__ gh,
                          const float* __restrict__ bi,
                          const float* __restrict__ bh,
                          const float* __restrict__ cinit,
                          float* __restrict__ out) {
  constexpr int PADK = 40;             // padded k-stride (elems) per n-row
  constexpr int GSTR = 16 * PADK;      // 640 elems per gate tile
  __shared__ __attribute__((aligned(16))) __bf16 smem[2][4 * GSTR];  // 10 KB

  const int tid    = threadIdx.x;
  const int lane   = tid & 31;
  const int wave   = tid >> 5;
  const int laneLo = lane & 15;
  const int laneHi = lane >> 4;

  const int mBase = blockIdx.x * 256 + wave * 32;   // 32 rows per wave
  const int hBase = blockIdx.y * 16;                // 16 h-cols per block

  const __bf16* Xb = (const __bf16*)Xp;
  const __bf16* Wb = (const __bf16*)Wp;

  // B staging role: thread -> (gate, n, k-chunk of 8), 16B each = 4KB/block
  const int sg = tid >> 6;            // 0..3
  const int sn = (tid >> 2) & 15;     // 0..15
  const int sq = tid & 3;             // 0..3
  const __bf16* wsrc = Wb + (size_t)(sg * kH + hBase + sn) * kI + sq * 8;
  const int sdst = sg * GSTR + sn * PADK + sq * 8;

  const size_t ar[2] = {(size_t)(mBase + laneLo) * kI,
                        (size_t)(mBase + 16 + laneLo) * kI};

  v8f acc[2][4];
#pragma unroll
  for (int mi = 0; mi < 2; ++mi)
#pragma unroll
    for (int g = 0; g < 4; ++g)
      acc[mi][g] = (v8f){0.f, 0.f, 0.f, 0.f, 0.f, 0.f, 0.f, 0.f};

  // prime buffer 0 with kb=0
  v8bf breg = *(const v8bf*)wsrc;
  *(v8bf*)(&smem[0][sdst]) = breg;
  __syncthreads();

  for (int i = 0; i < kI / 32; ++i) {
    const int kb  = i * 32;
    const int buf = i & 1;

    if (i < kI / 32 - 1)
      breg = *(const v8bf*)(wsrc + kb + 32);   // prefetch next B chunk

    __builtin_prefetch(Xb + ar[0] + kb + 128, 0, 0);  // global_prefetch_b8
    __builtin_prefetch(Xb + ar[1] + kb + 128, 0, 0);

    // B fragments from LDS (shared by all 8 waves)
    v16bf bfr[4];
#pragma unroll
    for (int g = 0; g < 4; ++g) {
      const __bf16* base = &smem[buf][g * GSTR + laneLo * PADK + laneHi * 16];
      bfr[g] = cat8(*(const v8bf*)base, *(const v8bf*)(base + 8));
    }

    // A fragments from global (per-wave rows, L2-resident)
    v16bf afr[2];
#pragma unroll
    for (int mi = 0; mi < 2; ++mi)
      afr[mi] = load_a_frag_bf16(Xb + ar[mi], kb, laneHi);

#pragma unroll
    for (int g = 0; g < 4; ++g) {
#pragma unroll
      for (int mi = 0; mi < 2; ++mi)
        acc[mi][g] = wmma_bf16(afr[mi], bfr[g], acc[mi][g]);
    }

    if (i < kI / 32 - 1) {
      *(v8bf*)(&smem[buf ^ 1][sdst]) = breg;   // fill other buffer
      __syncthreads();                         // readers done + fill visible
    }
  }

  lstm_epilogue(acc, mBase, hBase, laneLo, laneHi, gh, bi, bh, cinit, out);
}

// ------------- fallback GEMM (no/partial ws): direct global loads ----------
// Wave: 32 M-rows x 16 h-cols x 4 gates; block: 4 waves over h.
template <bool BBF>
__global__ __launch_bounds__(128, 1)
void lstm_main_kernel(const void* __restrict__ Xp,    // [B*T, I] f32
                      const void* __restrict__ Wp,    // [4H, I]  f32 or bf16
                      const float* __restrict__ gh,
                      const float* __restrict__ bi,
                      const float* __restrict__ bh,
                      const float* __restrict__ cinit,
                      float* __restrict__ out) {
  const int lane   = threadIdx.x & 31;
  const int wave   = threadIdx.x >> 5;
  const int laneLo = lane & 15;
  const int laneHi = lane >> 4;

  const int mBase = blockIdx.x * 32;
  const int hBase = blockIdx.y * 64 + wave * 16;

  const float*  Xf = (const float*)Xp;
  const float*  Wf = (const float*)Wp;
  const __bf16* Wb = (const __bf16*)Wp;

  const size_t ar[2] = {(size_t)(mBase + laneLo) * kI,
                        (size_t)(mBase + 16 + laneLo) * kI};
  size_t wr[4];
#pragma unroll
  for (int g = 0; g < 4; ++g)
    wr[g] = (size_t)(g * kH + hBase + laneLo) * kI;

  v8f acc[2][4];
#pragma unroll
  for (int mi = 0; mi < 2; ++mi)
#pragma unroll
    for (int g = 0; g < 4; ++g)
      acc[mi][g] = (v8f){0.f, 0.f, 0.f, 0.f, 0.f, 0.f, 0.f, 0.f};

  for (int kb = 0; kb < kI; kb += 32) {
    v16bf afr[2], bfr[4];
#pragma unroll
    for (int mi = 0; mi < 2; ++mi)
      afr[mi] = load_a_frag_f32(Xf + ar[mi], kb, laneHi);
    if constexpr (BBF) {
#pragma unroll
      for (int g = 0; g < 4; ++g)
        bfr[g] = load_b_frag_bf16(Wb + wr[g], kb, laneHi);
    } else {
#pragma unroll
      for (int g = 0; g < 4; ++g)
        bfr[g] = load_b_frag_f32(Wf + wr[g], kb, laneHi);
    }
#pragma unroll
    for (int g = 0; g < 4; ++g) {
#pragma unroll
      for (int mi = 0; mi < 2; ++mi)
        acc[mi][g] = wmma_bf16(afr[mi], bfr[g], acc[mi][g]);
    }
  }

  lstm_epilogue(acc, mBase, hBase, laneLo, laneHi, gh, bi, bh, cinit, out);
}

extern "C" void kernel_launch(void* const* d_in, const int* in_sizes, int n_in,
                              void* d_out, int out_size, void* d_ws, size_t ws_size,
                              hipStream_t stream) {
  (void)in_sizes; (void)n_in; (void)out_size;

  const float* X  = (const float*)d_in[0];  // inputs [B,T,I]
  const float* h0 = (const float*)d_in[1];  // [1,B,H]
  const float* c0 = (const float*)d_in[2];  // [1,B,H]
  const float* Wi = (const float*)d_in[3];  // [4,H,I]
  const float* Wh = (const float*)d_in[4];  // [4,H,H]
  const float* bi = (const float*)d_in[5];  // [4,H]
  const float* bh = (const float*)d_in[6];  // [4,H]
  float* out = (float*)d_out;

  char* ws = (char*)d_ws;
  const size_t ghBytes = (size_t)kB * kN * sizeof(float);     // 1 MB
  const size_t wiBytes = (size_t)kN * kI * 2;                 // 8 MB bf16
  const size_t xBytes  = (size_t)kB * kT * kI * 2;            // 64 MB bf16

  float* gh  = (float*)ws;
  void*  wiB = (void*)(ws + ghBytes);
  void*  xB  = (void*)(ws + ghBytes + wiBytes);

  const bool haveWi = ws_size >= ghBytes + wiBytes;
  const bool haveX  = ws_size >= ghBytes + wiBytes + xBytes;

  // tiny recurrent GEMM into ws
  gh_gemm_kernel<<<dim3(kB / 16, kN / 16), 32, 0, stream>>>(h0, Wh, gh);

  if (haveWi) {
    const int n8 = kN * kI / 8;
    f32_to_bf16_kernel<<<(n8 + 255) / 256, 256, 0, stream>>>(Wi, wiB, n8);
  }
  if (haveX) {
    const int n8 = kB * kT * kI / 8;
    f32_to_bf16_kernel<<<(n8 + 255) / 256, 256, 0, stream>>>(X, xB, n8);
  }

  if (haveX) {
    // 8 waves x 32 rows = 256 rows/block; 16 h-cols/block
    lstm_bf16_lds_kernel<<<dim3((kB * kT) / 256, kH / 16), 256, 0, stream>>>(
        xB, wiB, gh, bi, bh, c0, out);
  } else if (haveWi) {
    lstm_main_kernel<true><<<dim3((kB * kT) / 32, kH / 64), 128, 0, stream>>>(
        X, wiB, gh, bi, bh, c0, out);
  } else {
    lstm_main_kernel<false><<<dim3((kB * kT) / 32, kH / 64), 128, 0, stream>>>(
        X, Wi, gh, bi, bh, c0, out);
  }
}
